// LongformerSelfAttention_53996328846041
// MI455X (gfx1250) — matrix operand order repaired and checked by reading, not compiled, not run on hardware
//
#include <hip/hip_runtime.h>

// ---------------------------------------------------------------------------
// Types for CDNA5 WMMA (gfx1250, wave32)
// ---------------------------------------------------------------------------
typedef __bf16 bf16_t;
typedef bf16_t v16bf __attribute__((ext_vector_type(16)));
typedef bf16_t v8bf  __attribute__((ext_vector_type(8)));
typedef float  v8f   __attribute__((ext_vector_type(8)));

static __device__ __forceinline__ bf16_t f2bf(float f) {
  unsigned u = __builtin_bit_cast(unsigned, f);
  u += 0x7FFFu + ((u >> 16) & 1u);              // round-to-nearest-even
  unsigned short hs = (unsigned short)(u >> 16);
  return __builtin_bit_cast(bf16_t, hs);
}

static __device__ __forceinline__ v8f vzero8() {
  v8f z;
#pragma unroll
  for (int i = 0; i < 8; ++i) z[i] = 0.f;
  return z;
}

static __device__ __forceinline__ v16bf cat16(v8bf lo, v8bf hi) {
  return __builtin_shufflevector(lo, hi, 0,1,2,3,4,5,6,7,8,9,10,11,12,13,14,15);
}

// 16-bit A-matrix 16x32 lane layout: lane holds row m = lane%16;
// vector elems 0..15 = K kb+16*half+e (contiguous, matches load_b).
static __device__ __forceinline__ v16bf load_a(const bf16_t* rowptr, int half) {
  v8bf lo = *(const v8bf*)(rowptr + 16 * half);
  v8bf hi = *(const v8bf*)(rowptr + 16 * half + 8);
  return cat16(lo, hi);
}

// 16-bit B-matrix 32x16 lane layout: lane holds col n = lane%16; 16 contiguous
// K values at kb + 16*half.  With the operand stored [N,K] row-major (x @ W^T
// form), this is 32 contiguous bytes per lane.
static __device__ __forceinline__ v16bf load_b(const bf16_t* rowptr, int half) {
  v8bf lo = *(const v8bf*)(rowptr + 16 * half);
  v8bf hi = *(const v8bf*)(rowptr + 16 * half + 8);
  return cat16(lo, hi);
}

static __device__ __forceinline__ v8f wmma_bf16(v16bf a, v16bf b, v8f c) {
  // 8 args: (neg_a, A, neg_b, B, c_mod, C, reuse_a, reuse_b)
  return __builtin_amdgcn_wmma_f32_16x16x32_bf16(false, a, false, b, (short)0, c,
                                                 false, false);
}

// ---------------------------------------------------------------------------
// fp32 -> bf16 conversion, 8 elements / thread (16B loads+stores)
// ---------------------------------------------------------------------------
__global__ __launch_bounds__(256) void cvt_f32_to_bf16_x8(
    const float* __restrict__ in, bf16_t* __restrict__ out, int n8) {
  int i = blockIdx.x * 256 + threadIdx.x;
  if (i >= n8) return;
  const float4* p = (const float4*)in + (size_t)i * 2;
  float4 a = p[0], b = p[1];
  v8bf o;
  o[0] = f2bf(a.x); o[1] = f2bf(a.y); o[2] = f2bf(a.z); o[3] = f2bf(a.w);
  o[4] = f2bf(b.x); o[5] = f2bf(b.y); o[6] = f2bf(b.z); o[7] = f2bf(b.w);
  *(v8bf*)(out + (size_t)i * 8) = o;
}

// ---------------------------------------------------------------------------
// Ping-pong pipelined GEMM core for  [M,1024]bf16 @ W[N,1024]^T.
// Wave computes 16(M) x 64(N).  Two named fragment sets alternate so the
// rotation needs no register copies; loads for step k+1 are in flight while
// step k's 4 WMMAs execute.
// ---------------------------------------------------------------------------
#define GEMM_PIPELINED_LOOP(arow, brow, half, acc)                          \
  v16bf a0 = load_a(arow, half);                                            \
  v16bf b0_0 = load_b(brow[0], half), b0_1 = load_b(brow[1], half);         \
  v16bf b0_2 = load_b(brow[2], half), b0_3 = load_b(brow[3], half);         \
  for (int kb0 = 0; kb0 < 1024; kb0 += 64) {                                \
    int k1 = kb0 + 32;                                                      \
    int k2 = (kb0 + 64 < 1024) ? kb0 + 64 : 0; /* wrap: dead loads */       \
    __builtin_prefetch(arow + kb0 + 128, 0, 3);                             \
    v16bf a1 = load_a(arow + k1, half);                                     \
    v16bf b1_0 = load_b(brow[0] + k1, half);                                \
    v16bf b1_1 = load_b(brow[1] + k1, half);                                \
    v16bf b1_2 = load_b(brow[2] + k1, half);                                \
    v16bf b1_3 = load_b(brow[3] + k1, half);                                \
    acc[0] = wmma_bf16(a0, b0_0, acc[0]);                                   \
    acc[1] = wmma_bf16(a0, b0_1, acc[1]);                                   \
    acc[2] = wmma_bf16(a0, b0_2, acc[2]);                                   \
    acc[3] = wmma_bf16(a0, b0_3, acc[3]);                                   \
    a0 = load_a(arow + k2, half);                                           \
    b0_0 = load_b(brow[0] + k2, half);                                      \
    b0_1 = load_b(brow[1] + k2, half);                                      \
    b0_2 = load_b(brow[2] + k2, half);                                      \
    b0_3 = load_b(brow[3] + k2, half);                                      \
    acc[0] = wmma_bf16(a1, b1_0, acc[0]);                                   \
    acc[1] = wmma_bf16(a1, b1_1, acc[1]);                                   \
    acc[2] = wmma_bf16(a1, b1_2, acc[2]);                                   \
    acc[3] = wmma_bf16(a1, b1_3, acc[3]);                                   \
  }

// ---------------------------------------------------------------------------
// Fused QKV projection:  [8192,1024] @ W^T + b  for Wq,Wk,Wv (N = 3072 total)
// V output written transposed [B,H,dh,L] for the attention P*V GEMM.
// ---------------------------------------------------------------------------
__global__ __launch_bounds__(256) void qkv_gemm(
    const bf16_t* __restrict__ xb,
    const bf16_t* __restrict__ wq, const bf16_t* __restrict__ wk,
    const bf16_t* __restrict__ wv,
    const float* __restrict__ bq, const float* __restrict__ bk,
    const float* __restrict__ bv,
    bf16_t* __restrict__ qb, bf16_t* __restrict__ kb, bf16_t* __restrict__ vT) {
  int lane = threadIdx.x & 31, w = threadIdx.x >> 5;
  int half = lane >> 4, l16 = lane & 15;
  int n0 = blockIdx.x * 64;                 // [0, 3072)
  int m0 = blockIdx.y * 128 + w * 16;       // [0, 8192)
  int sel = n0 >> 10;                       // 0=q 1=k 2=v (block-uniform)
  int nW  = n0 & 1023;
  const bf16_t* W    = (sel == 0) ? wq : (sel == 1) ? wk : wv;
  const float*  bias = (sel == 0) ? bq : (sel == 1) ? bk : bv;

  v8f acc[4];
#pragma unroll
  for (int j = 0; j < 4; ++j) acc[j] = vzero8();

  const bf16_t* arow = xb + (size_t)(m0 + l16) * 1024;
  const bf16_t* brow[4];
#pragma unroll
  for (int j = 0; j < 4; ++j) brow[j] = W + (size_t)(nW + j * 16 + l16) * 1024;

  GEMM_PIPELINED_LOOP(arow, brow, half, acc)

#pragma unroll
  for (int j = 0; j < 4; ++j) {
    int nc = nW + j * 16 + l16;
    float bb = bias[nc];
#pragma unroll
    for (int r = 0; r < 8; ++r) {
      int m = m0 + r + 8 * half;            // C/D layout row
      bf16_t o = f2bf(acc[j][r] + bb);
      if (sel == 0) {
        qb[(size_t)m * 1024 + nc] = o;
      } else if (sel == 1) {
        kb[(size_t)m * 1024 + nc] = o;
      } else {
        int bi = m >> 12, l = m & 4095, head = nc >> 6, d = nc & 63;
        vT[((size_t)(bi * 16 + head) * 64 + d) * 4096 + l] = o;
      }
    }
  }
}

// ---------------------------------------------------------------------------
// Flash-style sliding-window attention.
// grid.x = B*C*H = 256 (one (b,c,h)); grid.y = 4 (128 queries / block);
// each wave: 16-query tile, online softmax over 16 key-blocks of 64.
// K / V fragments are gathered into register arrays first so their loads
// clause together and the 8 WMMAs issue back-to-back behind one wait.
// ---------------------------------------------------------------------------
__global__ __launch_bounds__(256) void attn_kernel(
    const bf16_t* __restrict__ qb, const bf16_t* __restrict__ kbuf,
    const bf16_t* __restrict__ vT, bf16_t* __restrict__ ao) {
  __shared__ bf16_t lds[8][16 * 64];        // wave-private P staging (bf16)

  int lane = threadIdx.x & 31, w = threadIdx.x >> 5;
  int half = lane >> 4, l16 = lane & 15;
  int bch = blockIdx.x;
  int h = bch & 15, c = (bch >> 4) & 7, b = bch >> 7;
  int m0 = blockIdx.y * 128 + w * 16;       // query row within chunk [0,512)
  int qtok0 = c * 512 + m0;
  const float scale = 0.125f;               // 1/sqrt(64)
  const float L2E = 1.44269504088896340736f;

  // Q A-fragments (dh=64 -> two K-steps of 32), kept in registers.
  const bf16_t* qrow = qb + ((size_t)(b * 4096 + qtok0 + l16)) * 1024 + h * 64;
  v16bf aq0 = load_a(qrow, half);
  v16bf aq1 = load_a(qrow + 32, half);

  v8f oacc[4];
  float run_m[8], run_l[8];
#pragma unroll
  for (int j = 0; j < 4; ++j) oacc[j] = vzero8();
#pragma unroll
  for (int r = 0; r < 8; ++r) { run_m[r] = -3.0e38f; run_l[r] = 0.f; }

  bf16_t* myl = &lds[w][0];
  int nb0 = (c == 0) ? 8 : 0;               // chunk 0: lookback fully masked

  for (int nb = nb0; nb < 16; ++nb) {
    int tkb = c * 512 - 512 + nb * 64;      // first key token of this block

    // ---- gather all K fragments, then S = Q K^T (16 x 64 tile) ----
    v16bf kf[8];
#pragma unroll
    for (int j = 0; j < 4; ++j) {
      int tk = tkb + j * 16 + l16;
      int tkc = tk < 0 ? 0 : tk;
      const bf16_t* krow = kbuf + ((size_t)(b * 4096 + tkc)) * 1024 + h * 64;
      kf[2 * j]     = load_b(krow, half);
      kf[2 * j + 1] = load_b(krow + 32, half);
    }
    v8f s[4];
#pragma unroll
    for (int j = 0; j < 4; ++j) {
      s[j] = vzero8();
      s[j] = wmma_bf16(aq0, kf[2 * j], s[j]);
      s[j] = wmma_bf16(aq1, kf[2 * j + 1], s[j]);
    }

    // ---- scale + mask ----
#pragma unroll
    for (int j = 0; j < 4; ++j) {
      bool valid = (tkb + j * 16 + l16) >= 0;
#pragma unroll
      for (int r = 0; r < 8; ++r) s[j][r] = valid ? s[j][r] * scale : -1.0e9f;
    }

    // ---- online-softmax row stats (rows live in 16-lane groups) ----
    float alpha[8];
#pragma unroll
    for (int r = 0; r < 8; ++r) {
      float m = fmaxf(fmaxf(s[0][r], s[1][r]), fmaxf(s[2][r], s[3][r]));
      m = fmaxf(m, __shfl_xor(m, 1));
      m = fmaxf(m, __shfl_xor(m, 2));
      m = fmaxf(m, __shfl_xor(m, 4));
      m = fmaxf(m, __shfl_xor(m, 8));
      float nm = fmaxf(run_m[r], m);
      alpha[r] = exp2f((run_m[r] - nm) * L2E);
      run_m[r] = nm;
    }

    float psum[8];
#pragma unroll
    for (int r = 0; r < 8; ++r) psum[r] = 0.f;
#pragma unroll
    for (int j = 0; j < 4; ++j)
#pragma unroll
      for (int r = 0; r < 8; ++r) {
        float p = exp2f((s[j][r] - run_m[r]) * L2E);
        s[j][r] = p;
        psum[r] += p;
      }
#pragma unroll
    for (int r = 0; r < 8; ++r) {
      float ps = psum[r];
      ps += __shfl_xor(ps, 1);
      ps += __shfl_xor(ps, 2);
      ps += __shfl_xor(ps, 4);
      ps += __shfl_xor(ps, 8);
      run_l[r] = run_l[r] * alpha[r] + ps;
    }
#pragma unroll
    for (int jo = 0; jo < 4; ++jo)
#pragma unroll
      for (int r = 0; r < 8; ++r) oacc[jo][r] *= alpha[r];

    // ---- C/D layout -> A layout through LDS (wave-private; DS in-order) ----
#pragma unroll
    for (int j = 0; j < 4; ++j)
#pragma unroll
      for (int r = 0; r < 8; ++r)
        myl[(r + 8 * half) * 64 + j * 16 + l16] = f2bf(s[j][r]);
    asm volatile("s_wait_dscnt 0" ::: "memory");

    const bf16_t* prow = myl + l16 * 64;
    v16bf pa0 = load_a(prow, half);         // keys 0..31 of block
    v16bf pa1 = load_a(prow + 32, half);    // keys 32..63 of block

    // ---- gather all V fragments, then O += P * V  (V is [B,H,dh,L]) ----
    v16bf vb[8];
#pragma unroll
    for (int jo = 0; jo < 4; ++jo) {
      const bf16_t* vrow =
          vT + ((size_t)(b * 16 + h) * 64 + jo * 16 + l16) * 4096;
#pragma unroll
      for (int kk2 = 0; kk2 < 2; ++kk2) {
        int tok = tkb + kk2 * 32 + 16 * half;   // 16-aligned: all-in or all-out
        int tokc = tok < 0 ? 0 : tok;
        v8bf lo = *(const v8bf*)(vrow + tokc);
        v8bf hi = *(const v8bf*)(vrow + tokc + 8);
        vb[2 * jo + kk2] = cat16(lo, hi);
      }
    }
#pragma unroll
    for (int jo = 0; jo < 4; ++jo) {
      oacc[jo] = wmma_bf16(pa0, vb[2 * jo], oacc[jo]);
      oacc[jo] = wmma_bf16(pa1, vb[2 * jo + 1], oacc[jo]);
    }
  }

  // ---- normalize + store (bf16 for the output projection) ----
#pragma unroll
  for (int jo = 0; jo < 4; ++jo) {
    int col = h * 64 + jo * 16 + l16;
#pragma unroll
    for (int r = 0; r < 8; ++r) {
      int qt = qtok0 + r + 8 * half;
      float o = oacc[jo][r] / run_l[r];
      ao[((size_t)(b * 4096 + qt)) * 1024 + col] = f2bf(o);
    }
  }
}

// ---------------------------------------------------------------------------
// Output projection: ao_bf16 @ Wo^T + bo -> fp32 d_out (ping-pong pipelined)
// ---------------------------------------------------------------------------
__global__ __launch_bounds__(256) void oproj_gemm(
    const bf16_t* __restrict__ ab, const bf16_t* __restrict__ wo,
    const float* __restrict__ bo, float* __restrict__ out) {
  int lane = threadIdx.x & 31, w = threadIdx.x >> 5;
  int half = lane >> 4, l16 = lane & 15;
  int n0 = blockIdx.x * 64;
  int m0 = blockIdx.y * 128 + w * 16;

  v8f acc[4];
#pragma unroll
  for (int j = 0; j < 4; ++j) acc[j] = vzero8();

  const bf16_t* arow = ab + (size_t)(m0 + l16) * 1024;
  const bf16_t* brow[4];
#pragma unroll
  for (int j = 0; j < 4; ++j) brow[j] = wo + (size_t)(n0 + j * 16 + l16) * 1024;

  GEMM_PIPELINED_LOOP(arow, brow, half, acc)

#pragma unroll
  for (int j = 0; j < 4; ++j) {
    int nc = n0 + j * 16 + l16;
    float bb = bo[nc];
#pragma unroll
    for (int r = 0; r < 8; ++r) {
      int m = m0 + r + 8 * half;
      out[(size_t)m * 1024 + nc] = acc[j][r] + bb;
    }
  }
}

// ---------------------------------------------------------------------------
// Host launcher
// ---------------------------------------------------------------------------
extern "C" void kernel_launch(void* const* d_in, const int* in_sizes, int n_in,
                              void* d_out, int out_size, void* d_ws,
                              size_t ws_size, hipStream_t stream) {
  (void)in_sizes; (void)n_in; (void)out_size; (void)ws_size;
  const float* x  = (const float*)d_in[0];
  const float* Wq = (const float*)d_in[1];
  const float* bq = (const float*)d_in[2];
  const float* Wk = (const float*)d_in[3];
  const float* bk = (const float*)d_in[4];
  const float* Wv = (const float*)d_in[5];
  const float* bv = (const float*)d_in[6];
  const float* Wo = (const float*)d_in[7];
  const float* bo = (const float*)d_in[8];
  float* out = (float*)d_out;

  const size_t MB = 1ull << 20;
  char* ws = (char*)d_ws;
  bf16_t* xb   = (bf16_t*)(ws + 0 * MB);    // 16 MB  x bf16 [8192,1024]
  bf16_t* wqb  = (bf16_t*)(ws + 16 * MB);   //  2 MB
  bf16_t* wkb  = (bf16_t*)(ws + 18 * MB);   //  2 MB
  bf16_t* wvb  = (bf16_t*)(ws + 20 * MB);   //  2 MB
  bf16_t* wob  = (bf16_t*)(ws + 22 * MB);   //  2 MB
  bf16_t* qbuf = (bf16_t*)(ws + 24 * MB);   // 16 MB  q bf16 [8192,1024]
  bf16_t* kbuf = (bf16_t*)(ws + 40 * MB);   // 16 MB  k bf16 [8192,1024]
  bf16_t* vT   = (bf16_t*)(ws + 56 * MB);   // 16 MB  v bf16 [B,H,64,4096]
  bf16_t* ao   = (bf16_t*)(ws + 72 * MB);   // 16 MB  attn out bf16 [8192,1024]

  // Convert inputs to bf16
  cvt_f32_to_bf16_x8<<<4096, 256, 0, stream>>>(x,  xb,  (8192 * 1024) / 8);
  cvt_f32_to_bf16_x8<<<512,  256, 0, stream>>>(Wq, wqb, (1024 * 1024) / 8);
  cvt_f32_to_bf16_x8<<<512,  256, 0, stream>>>(Wk, wkb, (1024 * 1024) / 8);
  cvt_f32_to_bf16_x8<<<512,  256, 0, stream>>>(Wv, wvb, (1024 * 1024) / 8);
  cvt_f32_to_bf16_x8<<<512,  256, 0, stream>>>(Wo, wob, (1024 * 1024) / 8);

  // QKV projections (N = 3*1024, M = 8192)
  qkv_gemm<<<dim3(48, 64), 256, 0, stream>>>(xb, wqb, wkb, wvb, bq, bk, bv,
                                             qbuf, kbuf, vT);
  // Sliding-window attention (B*C*H = 256, 4 query blocks of 128)
  attn_kernel<<<dim3(256, 4), 256, 0, stream>>>(qbuf, kbuf, vT, ao);
  // Output projection
  oproj_gemm<<<dim3(16, 64), 256, 0, stream>>>(ao, wob, bo, out);
}
